// Head_12283606468094
// MI455X (gfx1250) — matrix operand order repaired and checked by reading, not compiled, not run on hardware
//
#include <hip/hip_runtime.h>

typedef __attribute__((ext_vector_type(16))) _Float16 v16h;
typedef __attribute__((ext_vector_type(8)))  _Float16 v8h;
typedef __attribute__((ext_vector_type(8)))  float    v8f;
typedef __attribute__((ext_vector_type(4)))  float    v4f;

static constexpr int Bn  = 64;
static constexpr int Tn  = 256;
static constexpr int Cn  = 2048;
static constexpr int HSn = 128;

// ---------------------------------------------------------------------------
// Kernel 1: fused projections.  K,Q stored row-major [B*T, HS] f16,
// V stored transposed [B, HS, T] f16 so attention B-fragments are contiguous.
// Block = 128 threads (4 waves). Each block: a 32-row tile of B*T, one of
// {Wk,Wq,Wv} (blockIdx.y). Wave w: m-tile = w&1, n-half = (w>>1)*64.
// Fragments are prefetched into registers before the WMMA chain so LDS
// loads issue as one clause and the matrix pipe runs back-to-back.
// ---------------------------------------------------------------------------
__global__ __launch_bounds__(128)
void proj_kernel(const float* __restrict__ x,
                 const float* __restrict__ Wk,
                 const float* __restrict__ Wq,
                 const float* __restrict__ Wv,
                 _Float16* __restrict__ kf,
                 _Float16* __restrict__ qf,
                 _Float16* __restrict__ vT)
{
    constexpr int XP = 40;   // padded row stride (halves) for x tile in LDS
    constexpr int WP = 40;   // padded row stride (halves) for W^T tile in LDS
    __shared__ _Float16 xs[32 * XP];
    __shared__ _Float16 wsT[128 * WP];

    const int tid  = threadIdx.x;
    const int lane = tid & 31;
    const int wave = tid >> 5;      // 0..3
    const int li   = lane & 15;
    const int hi   = lane >> 4;
    const int mt   = wave & 1;      // 16-row m-tile within the 32-row block
    const int nh   = wave >> 1;     // 64-col n-half

    const int which = blockIdx.y;   // 0=Wk 1=Wq 2=Wv
    const float* W = (which == 0) ? Wk : (which == 1) ? Wq : Wv;

    const size_t gRow0 = (size_t)blockIdx.x * 32;

    v8f acc[4] = {};

    for (int k0 = 0; k0 < Cn; k0 += 32) {
        // ---- stage x tile [32 x 32] fp32 -> f16 LDS (8 elems/thread) ----
        {
            int idx = tid * 8;
            int r = idx >> 5;
            int c = idx & 31;
            const float* xp = x + (gRow0 + r) * Cn + k0 + c;
            v4f v0 = *(const v4f*)xp;
            v4f v1 = *(const v4f*)(xp + 4);
            _Float16* dst = &xs[r * XP + c];
#pragma unroll
            for (int j = 0; j < 4; ++j) { dst[j] = (_Float16)v0[j]; dst[4 + j] = (_Float16)v1[j]; }
        }
        // ---- stage W tile [32 x 128] transposed into LDS (thread = column) ----
        {
            const float* wp = W + (size_t)k0 * HSn + tid;
#pragma unroll
            for (int kk = 0; kk < 32; ++kk)
                wsT[tid * WP + kk] = (_Float16)wp[kk * HSn];
        }
        __syncthreads();

        // ---- prefetch A fragment + all 4 B fragments, then WMMA chain ----
        v16h a;
        {
            const _Float16* ar = &xs[(mt * 16 + li) * XP];
#pragma unroll
            for (int j = 0; j < 8; ++j) {
                a[j]     = ar[hi * 8 + j];
                a[8 + j] = ar[16 + hi * 8 + j];
            }
        }
        v16h bf[4];
#pragma unroll
        for (int j2 = 0; j2 < 4; ++j2) {
            int n = nh * 64 + j2 * 16 + li;
#pragma unroll
            for (int i = 0; i < 16; ++i) bf[j2][i] = wsT[n * WP + (i < 8 ? hi * 8 + i : 16 + hi * 8 + (i - 8))];
        }
#pragma unroll
        for (int j2 = 0; j2 < 4; ++j2)
            acc[j2] = __builtin_amdgcn_wmma_f32_16x16x32_f16(
                false, a, false, bf[j2], (short)0, acc[j2], false, false);
        __syncthreads();
    }

    // ---- epilogue: uniform branch hoisted; Vt tiles stored contiguously ----
    if (which == 2) {
        // vT[(b*HS + n)*T + t]; acc rows r map to consecutive t -> v8h store
        size_t b  = gRow0 >> 8;                       // /T (tiles never straddle)
        size_t t0 = (gRow0 & 255) + mt * 16 + hi * 8; // + r is contiguous
#pragma unroll
        for (int j2 = 0; j2 < 4; ++j2) {
            int n = nh * 64 + j2 * 16 + li;
            v8h pack;
#pragma unroll
            for (int r = 0; r < 8; ++r) pack[r] = (_Float16)acc[j2][r];
            *(v8h*)(vT + (b * HSn + n) * Tn + t0) = pack;
        }
    } else {
        _Float16* __restrict__ dst = which ? qf : kf;
        size_t g0 = gRow0 + mt * 16 + hi * 8;
#pragma unroll
        for (int j2 = 0; j2 < 4; ++j2) {
            int n = nh * 64 + j2 * 16 + li;
#pragma unroll
            for (int r = 0; r < 8; ++r)
                dst[(g0 + r) * HSn + n] = (_Float16)acc[j2][r];
        }
    }
}

// ---------------------------------------------------------------------------
// Kernel 2: causal attention. One wave per (batch, 16-row t-tile).
// scores = k @ q^T * C^-0.5, causal softmax, out = weights @ v.
// ---------------------------------------------------------------------------
__global__ __launch_bounds__(32)
void attn_kernel(const _Float16* __restrict__ kf,
                 const _Float16* __restrict__ qf,
                 const _Float16* __restrict__ vT,
                 float* __restrict__ out)
{
    __shared__ float    sS[16 * Tn];   // scaled+masked scores (f32)
    __shared__ _Float16 sP[16 * Tn];   // unnormalized exp weights (f16)
    __shared__ float    sRS[16];       // 1/rowsum

    const int lane = threadIdx.x;
    const int li   = lane & 15;
    const int hi   = lane >> 4;
    const int tt   = blockIdx.x;           // t-tile 0..15
    const int b    = blockIdx.y;           // batch
    const int tbase  = tt * 16;
    const int sLimit = (tt + 1) * 16;      // causal extent

    // zero weight buffer so the 32-wide K tail of GEMM2 reads zeros
#pragma unroll
    for (int j = 0; j < 128; ++j) sP[lane * 128 + j] = (_Float16)0.f;

    const _Float16* kb = kf + (size_t)(b * Tn) * HSn;
    const _Float16* qb = qf + (size_t)(b * Tn) * HSn;
    const _Float16* vb = vT + (size_t)(b * HSn) * Tn;

    const float scale = 0.022097086912079608f;   // 2048^-0.5

    // ---- prefetch all 4 A fragments (k rows, reused for every ss) ----
    v16h afr[4];
#pragma unroll
    for (int kc = 0; kc < 4; ++kc) {
        const _Float16* ap = kb + (size_t)(tbase + li) * HSn + kc * 32;
        v8h a0 = *(const v8h*)(ap + hi * 16);
        v8h a1 = *(const v8h*)(ap + hi * 16 + 8);
#pragma unroll
        for (int j = 0; j < 8; ++j) { afr[kc][j] = a0[j]; afr[kc][8 + j] = a1[j]; }
    }

    // ---- scores: only tiles on/below the diagonal ----
    for (int ss = 0; ss <= tt; ++ss) {
        // prefetch all 4 q^T B-fragments as one load clause
        v16h bfr[4];
#pragma unroll
        for (int kc = 0; kc < 4; ++kc) {
            const _Float16* bp = qb + (size_t)(ss * 16 + li) * HSn + kc * 32 + hi * 16;
            bfr[kc] = *(const v16h*)bp;
        }
        v8f acc = {};
#pragma unroll
        for (int kc = 0; kc < 4; ++kc)
            acc = __builtin_amdgcn_wmma_f32_16x16x32_f16(
                false, afr[kc], false, bfr[kc], (short)0, acc, false, false);
#pragma unroll
        for (int r = 0; r < 8; ++r) {
            int m = hi * 8 + r;
            int s = ss * 16 + li;
            float v = acc[r] * scale;
            if (ss == tt && s > tbase + m) v = -__builtin_inff();
            sS[m * Tn + s] = v;
        }
    }
    asm volatile("s_wait_dscnt 0" ::: "memory");

    // ---- softmax: lane half hi scans half of [0, sLimit) for row li ----
    {
        const int row = li;
        const int halfLen = sLimit >> 1;
        const int s0 = hi * halfLen;
        float mx = -__builtin_inff();
        for (int s = s0; s < s0 + halfLen; ++s)
            mx = fmaxf(mx, sS[row * Tn + s]);
        mx = fmaxf(mx, __shfl_xor(mx, 16));
        float sum = 0.f;
        for (int s = s0; s < s0 + halfLen; ++s) {
            float e = exp2f((sS[row * Tn + s] - mx) * 1.4426950408889634f);
            sum += e;
            sP[row * Tn + s] = (_Float16)e;
        }
        sum += __shfl_xor(sum, 16);
        if (hi == 0) sRS[row] = 1.0f / sum;
    }
    asm volatile("s_wait_dscnt 0" ::: "memory");

    // ---- O = E @ V (K runs over s in chunks of 32, causal-truncated) ----
    v8f o[8];
#pragma unroll
    for (int nt = 0; nt < 8; ++nt) o[nt] = (v8f){};
    const int kcN = (sLimit + 31) >> 5;
    for (int kc = 0; kc < kcN; ++kc) {
        int s0 = kc * 32;
        v16h a;
#pragma unroll
        for (int j = 0; j < 8; ++j) {
            a[j]     = sP[li * Tn + s0 + hi * 16 + j];
            a[8 + j] = sP[li * Tn + s0 + hi * 16 + 8 + j];
        }
        // prefetch all 8 V^T B-fragments (one global clause), then 8 WMMAs
        v16h bfr[8];
#pragma unroll
        for (int nt = 0; nt < 8; ++nt) {
            const _Float16* bp = vb + (size_t)(nt * 16 + li) * Tn + s0 + hi * 16;
            bfr[nt] = *(const v16h*)bp;
        }
#pragma unroll
        for (int nt = 0; nt < 8; ++nt)
            o[nt] = __builtin_amdgcn_wmma_f32_16x16x32_f16(
                false, a, false, bfr[nt], (short)0, o[nt], false, false);
    }

    // ---- epilogue: fold in 1/rowsum, write fp32 output ----
    float rsv[8];
#pragma unroll
    for (int r = 0; r < 8; ++r) rsv[r] = sRS[hi * 8 + r];
#pragma unroll
    for (int nt = 0; nt < 8; ++nt) {
#pragma unroll
        for (int r = 0; r < 8; ++r) {
            int m = hi * 8 + r;
            int h = nt * 16 + li;
            out[((size_t)(b * Tn + tbase + m)) * HSn + h] = o[nt][r] * rsv[r];
        }
    }
}

// ---------------------------------------------------------------------------
extern "C" void kernel_launch(void* const* d_in, const int* in_sizes, int n_in,
                              void* d_out, int out_size, void* d_ws, size_t ws_size,
                              hipStream_t stream)
{
    const float* x  = (const float*)d_in[0];
    const float* Wk = (const float*)d_in[1];
    const float* Wq = (const float*)d_in[2];
    const float* Wv = (const float*)d_in[3];
    float* out = (float*)d_out;

    // workspace: kf16 | qf16 | vT16  (each B*T*HS halves = 4 MiB, 12 MiB total)
    _Float16* kf = (_Float16*)d_ws;
    _Float16* qf = kf + (size_t)Bn * Tn * HSn;
    _Float16* vT = qf + (size_t)Bn * Tn * HSn;

    dim3 g1(Bn * Tn / 32, 3), b1(128);
    proj_kernel<<<g1, b1, 0, stream>>>(x, Wk, Wq, Wv, kf, qf, vT);

    dim3 g2(Tn / 16, Bn), b2(32);
    attn_kernel<<<g2, b2, 0, stream>>>(kf, qf, vT, out);
}
